// GCN_EAI_27642409517527
// MI455X (gfx1250) — matrix operand order repaired and checked by reading, not compiled, not run on hardware
//
#include <hip/hip_runtime.h>
#include <hip/hip_bf16.h>
#include <stdint.h>

// Problem dims (from reference)
#define BB   8
#define SSQ  1024
#define EE   256
#define HH   8
#define HDD  32
#define MT   (BB*SSQ)    // 8192 total rows
#define QN   (3*EE)      // 768 qkv columns
#define QSCALE 0.17677669529663689f  // 32^-0.5
#define EPSV 1e-5f

typedef __attribute__((ext_vector_type(16))) _Float16 v16h;
typedef __attribute__((ext_vector_type(8)))  _Float16 v8h;
typedef __attribute__((ext_vector_type(8)))  float    v8f;
typedef __attribute__((ext_vector_type(4)))  unsigned int u32x4;
typedef __attribute__((ext_vector_type(8)))  int          i32x8;
typedef __attribute__((ext_vector_type(4)))  int          i32x4;

// ------------------------------------------------- DPP 16-lane reductions
// Butterfly over XOR masks {1,2,7,15} (linearly independent over GF(2)):
// quad_perm xor1 (0xB1), quad_perm xor2 (0x4E), row_half_mirror (0x141 = xor7),
// row_mirror (0x140 = xor15).  Pure VALU: co-executes with WMMA, no DS traffic.
template<int CTRL>
__device__ __forceinline__ float dppf(float x) {
    return __int_as_float(__builtin_amdgcn_update_dpp(
        0, __float_as_int(x), CTRL, 0xF, 0xF, true));
}
__device__ __forceinline__ float rowmax16(float v) {
    v = fmaxf(v, dppf<0xB1>(v));
    v = fmaxf(v, dppf<0x4E>(v));
    v = fmaxf(v, dppf<0x141>(v));
    v = fmaxf(v, dppf<0x140>(v));
    return v;
}
__device__ __forceinline__ float rowsum16(float v) {
    v += dppf<0xB1>(v);
    v += dppf<0x4E>(v);
    v += dppf<0x141>(v);
    v += dppf<0x140>(v);
    return v;
}

// ------------------------------------------------- Tensor Data Mover
// 2-D tile load: 32 rows x 32 f16 (64B) with row stride QN halves, into LDS.
// D# per ISA ch.8: group0={count=1, lds_addr, global_addr[56:0], type=2},
// group1={data_size=2B, tensor_dim=32x32, tile_dim=32x32, dim0_stride=QN}.
// Groups 2/3 (+ extra group in the 6-arg builtin) are only used for >2-D
// tensors -> zero.
__device__ __forceinline__ void tdm_load_tile(uint32_t ldsAddr, const void* gptr) {
    uint64_t ga = (uint64_t)(uintptr_t)gptr;
    u32x4 g0 = { 1u,                                    // count=1 descriptor
                 ldsAddr,                               // lds_addr (bytes)
                 (uint32_t)ga,                          // global_addr[31:0]
                 (uint32_t)(((ga >> 32) & 0x01FFFFFFull) | 0x80000000u) }; // [56:32] | type=2
    i32x8 g1 = { 0x00010000,          // wg_mask=0 | data_size=1 (2 bytes)
                 0x00200000,          // atomic_barrier=0 | tensor_dim0.lo=32
                 0x00200000,          // tensor_dim0.hi=0 | tensor_dim1.lo=32
                 0x00200000,          // tensor_dim1.hi=0 | tile_dim0=32
                 0x00000020,          // tile_dim1=32 | tile_dim2=0
                 (int)QN,             // tensor_dim0_stride = 768 elements
                 0, 0 };
    i32x4 z4 = { 0, 0, 0, 0 };
    i32x8 z8 = { 0, 0, 0, 0, 0, 0, 0, 0 };
    __builtin_amdgcn_tensor_load_to_lds(g0, g1, z4, z4, z8, 0);
}

// ---------------------------------------------------------------- convert
__global__ __launch_bounds__(256)
void f32_to_f16_kernel(const float* __restrict__ src, _Float16* __restrict__ dst, int n) {
    int i = blockIdx.x * blockDim.x + threadIdx.x;
    if (i < n) dst[i] = (_Float16)src[i];
}

// ---------------------------------------------------------------- GEMM
// C[M,N] = A[M,K] @ W[N,K]^T + bias.  Wave tile = 32(M) x 16(N): two WMMAs
// share one B fragment, halving weight traffic per FLOP.  Block = 4 waves
// covering 32(M) x 64(N).
template<bool QKV_MODE>
__global__ __launch_bounds__(128)
void wmma_gemm_kernel(const _Float16* __restrict__ A, const _Float16* __restrict__ W,
                      const float* __restrict__ bias, void* __restrict__ Cout,
                      int M, int N, int K, float qscale, int qcols)
{
    const int wave = threadIdx.x >> 5;
    const int lane = threadIdx.x & 31;
    const int lh   = lane & 15;
    const int sel  = lane >> 4;
    const int m0   = blockIdx.x * 32;
    const int n0   = blockIdx.y * 64 + wave * 16;

    const _Float16* Ar0 = A + (size_t)(m0 + lh) * K;        // A-frag rows m0..m0+15
    const _Float16* Ar1 = A + (size_t)(m0 + 16 + lh) * K;   // A-frag rows m0+16..m0+31
    const _Float16* Wr  = W + (size_t)(n0 + lh) * K;        // B-frag: N = lh
    __builtin_prefetch(Wr, 0, 1);   // global_prefetch_b8: weights resident in L2

    v8f acc0 = {}, acc1 = {};
    for (int k = 0; k < K; k += 32) {
        // A 16x32 f16 layout: lanes 0-15 K=0..15, lanes 16-31 K=16..31 (matches B)
        const v8h a0lo = *(const v8h*)(Ar0 + k + sel * 16);
        const v8h a0hi = *(const v8h*)(Ar0 + k + sel * 16 + 8);
        const v8h a1lo = *(const v8h*)(Ar1 + k + sel * 16);
        const v8h a1hi = *(const v8h*)(Ar1 + k + sel * 16 + 8);
        v16h a0, a1;
        #pragma unroll
        for (int i = 0; i < 8; ++i) {
            a0[i] = a0lo[i]; a0[i + 8] = a0hi[i];
            a1[i] = a1lo[i]; a1[i + 8] = a1hi[i];
        }
        // B 32x16 f16 layout: lanes 0-15 K=0..15, lanes 16-31 K=16..31, N = lh
        const v16h b = *(const v16h*)(Wr + k + sel * 16);
        acc0 = __builtin_amdgcn_wmma_f32_16x16x32_f16(false, a0, false, b, (short)0, acc0, false, false);
        acc1 = __builtin_amdgcn_wmma_f32_16x16x32_f16(false, a1, false, b, (short)0, acc1, false, false);
    }

    const int col = n0 + lh;
    const float bval = bias[col];
    if (QKV_MODE) {
        const float sc = (col < qcols) ? qscale : 1.0f;
        _Float16* C = (_Float16*)Cout;
        #pragma unroll
        for (int r = 0; r < 8; ++r) {   // C/D layout: M = r + sel*8, N = lh
            size_t row = (size_t)(m0 + r + sel * 8);
            C[row * (size_t)N + col]        = (_Float16)((acc0[r] + bval) * sc);
            C[(row + 16) * (size_t)N + col] = (_Float16)((acc1[r] + bval) * sc);
        }
    } else {
        float* C = (float*)Cout;
        #pragma unroll
        for (int r = 0; r < 8; ++r) {
            size_t row = (size_t)(m0 + r + sel * 8);
            C[row * (size_t)N + col]        = acc0[r] + bval;
            C[(row + 16) * (size_t)N + col] = acc1[r] + bval;
        }
    }
}

// ---------------------------------------------------------------- attention
// Flash-style: block = 8 waves, each wave owns 16 queries of one (b,h).
// K/V tiles (32 keys x 32 dims f16) are double-buffered in LDS and fetched by
// the Tensor Data Mover (wave 0 issues 2 descriptors/chunk; TENSORcnt is
// in-order so s_wait_tensorcnt(2) after issuing the prefetch guarantees the
// current chunk has landed).
__global__ __launch_bounds__(256)
void attn_kernel(const _Float16* __restrict__ qkv, const float* __restrict__ pde,
                 _Float16* __restrict__ att)
{
    __shared__ _Float16 lK[2][32 * 32];   // double-buffered K tiles
    __shared__ _Float16 lV[2][32 * 32];   // double-buffered V tiles
    __shared__ _Float16 lP[8][16 * 32];   // per-wave P re-striping buffer

    const int blk  = blockIdx.x;          // B*H*(S/128) = 512
    const int mBlk = blk & 7;
    const int h    = (blk >> 3) & 7;
    const int b    = blk >> 6;
    const int tid  = threadIdx.x;
    const int wave = tid >> 5;
    const int lane = tid & 31;
    const int lh   = lane & 15;
    const int sel  = lane >> 4;
    const int m0   = mBlk * 128 + wave * 16;
    const size_t rowBase = (size_t)b * SSQ;

    // Q fragment (A layout, K = HD = 32), q already scaled in the QKV GEMM
    const _Float16* qrow = qkv + (rowBase + m0 + lh) * QN + h * HDD;
    v16h qa;
    #pragma unroll
    for (int i = 0; i < 8; ++i) { qa[i] = qrow[sel * 16 + i]; qa[i + 8] = qrow[sel * 16 + 8 + i]; }

    v8f O0 = {}, O1 = {};
    float mi[8], li[8];
    #pragma unroll
    for (int r = 0; r < 8; ++r) { mi[r] = -1e30f; li[r] = 0.0f; }

    _Float16* lPw = &lP[wave][0];

    const _Float16* kBase = qkv + rowBase * QN + EE + h * HDD;      // K block cols
    const _Float16* vBase = qkv + rowBase * QN + 2 * EE + h * HDD;  // V block cols
    const uint32_t lkA[2] = { (uint32_t)(uintptr_t)&lK[0][0], (uint32_t)(uintptr_t)&lK[1][0] };
    const uint32_t lvA[2] = { (uint32_t)(uintptr_t)&lV[0][0], (uint32_t)(uintptr_t)&lV[1][0] };

    if (wave == 0) {                      // prime buffer 0
        tdm_load_tile(lkA[0], kBase);
        tdm_load_tile(lvA[0], vBase);
    }

    for (int kc = 0; kc < SSQ / 32; ++kc) {
        const int cur = kc & 1;
        if (wave == 0 && kc + 1 < SSQ / 32) {
            // prefetch next chunk into the other buffer, then wait for current
            tdm_load_tile(lkA[cur ^ 1], kBase + (size_t)(kc + 1) * 32 * QN);
            tdm_load_tile(lvA[cur ^ 1], vBase + (size_t)(kc + 1) * 32 * QN);
            __builtin_amdgcn_s_wait_tensorcnt((short)2);
        } else {
            __builtin_amdgcn_s_wait_tensorcnt((short)0);
        }
        __syncthreads();                  // current K/V tile visible to all waves
        const _Float16* Kc = lK[cur];
        const _Float16* Vc = lV[cur];

        // scores: two 16-key tiles, one WMMA each (K dim = HD = 32)
        v16h kb0, kb1;
        #pragma unroll
        for (int j = 0; j < 16; ++j) {    // B layout: col = key = lh, K = d
            kb0[j] = Kc[lh * 32 + sel * 16 + j];
            kb1[j] = Kc[(16 + lh) * 32 + sel * 16 + j];
        }
        v8f z = {};
        v8f s0 = __builtin_amdgcn_wmma_f32_16x16x32_f16(false, qa, false, kb0, (short)0, z, false, false);
        v8f s1 = __builtin_amdgcn_wmma_f32_16x16x32_f16(false, qa, false, kb1, (short)0, z, false, false);

        const float* pr = pde + (size_t)(m0 + sel * 8) * SSQ + (size_t)kc * 32;
        #pragma unroll
        for (int r = 0; r < 8; ++r) {
            float a0 = s0[r] + pr[(size_t)r * SSQ + lh];
            float a1 = s1[r] + pr[(size_t)r * SSQ + 16 + lh];
            // row-wise online softmax; DPP butterfly within each 16-lane half
            float mx = rowmax16(fmaxf(a0, a1));
            float mn = fmaxf(mi[r], mx);
            float fr = __expf(mi[r] - mn);
            float p0 = __expf(a0 - mn);
            float p1 = __expf(a1 - mn);
            float rs = rowsum16(p0 + p1);
            li[r] = li[r] * fr + rs;
            mi[r] = mn;
            O0[r] *= fr; O1[r] *= fr;
            // park P (C layout) into LDS row-major so it can be re-read in A layout
            lPw[(r + sel * 8) * 32 + lh]      = (_Float16)p0;
            lPw[(r + sel * 8) * 32 + 16 + lh] = (_Float16)p1;
        }
        asm volatile("s_wait_dscnt 0" ::: "memory");  // P stores -> P loads (same wave)

        v16h pa, vb0, vb1;
        #pragma unroll
        for (int i = 0; i < 8; ++i) {     // A layout read of P (16x32, K = key)
            pa[i]     = lPw[lh * 32 + sel * 16 + i];
            pa[i + 8] = lPw[lh * 32 + sel * 16 + 8 + i];
        }
        #pragma unroll
        for (int j = 0; j < 16; ++j) {    // V B-frags: N = dim halves 0-15 / 16-31
            vb0[j] = Vc[(sel * 16 + j) * 32 + lh];
            vb1[j] = Vc[(sel * 16 + j) * 32 + 16 + lh];
        }
        O0 = __builtin_amdgcn_wmma_f32_16x16x32_f16(false, pa, false, vb0, (short)0, O0, false, false);
        O1 = __builtin_amdgcn_wmma_f32_16x16x32_f16(false, pa, false, vb1, (short)0, O1, false, false);

        __syncthreads();                  // all reads done before TDM overwrites buffer
    }

    #pragma unroll
    for (int r = 0; r < 8; ++r) {
        float inv = 1.0f / li[r];
        size_t row = rowBase + m0 + r + sel * 8;
        att[row * EE + h * HDD + lh]      = (_Float16)(O0[r] * inv);
        att[row * EE + h * HDD + 16 + lh] = (_Float16)(O1[r] * inv);
    }
}

// ---------------------------------------------------------------- BN + MMD
__global__ __launch_bounds__(256)
void zero_kernel(float* p, int n) {
    int i = blockIdx.x * blockDim.x + threadIdx.x;
    if (i < n) p[i] = 0.0f;
}

// stats: [0:256)=sumY [256:512)=sumYsq [512:768)=sumY2 [768:1024)=sumY2sq
__global__ __launch_bounds__(256)
void bn_stats_kernel(const float* __restrict__ Y, const float* __restrict__ Y2,
                     float* __restrict__ stats)
{
    const int e = threadIdx.x;
    const size_t r0 = (size_t)blockIdx.x * 32;
    float s1 = 0, q1 = 0, s2 = 0, q2 = 0;
    for (int i = 0; i < 32; ++i) {
        float a = Y[(r0 + i) * EE + e];  s1 += a; q1 += a * a;
        float c = Y2[(r0 + i) * EE + e]; s2 += c; q2 += c * c;
    }
    atomicAdd(&stats[e],       s1);
    atomicAdd(&stats[256 + e], q1);
    atomicAdd(&stats[512 + e], s2);
    atomicAdd(&stats[768 + e], q2);
}

__global__ __launch_bounds__(256)
void bn_apply_kernel(float* __restrict__ Y, float* __restrict__ Y2,
                     const float* __restrict__ stats,
                     const float* __restrict__ g1, const float* __restrict__ b1,
                     const float* __restrict__ g2, const float* __restrict__ b2,
                     const float* __restrict__ bnw,
                     float* __restrict__ rmY, float* __restrict__ rmY2)
{
    const int e = threadIdx.x;
    const size_t row = blockIdx.x;
    const float invN = 1.0f / (float)MT;
    float m1 = stats[e] * invN;
    float v1 = stats[256 + e] * invN - m1 * m1;
    float m2 = stats[512 + e] * invN;
    float v2 = stats[768 + e] * invN - m2 * m2;
    float w = (1.0f / (1.0f + __expf(-bnw[0])) + 1.0f) * 0.5f;
    float mfa = w * m1 + (1 - w) * m2, mfb = w * m2 + (1 - w) * m1;
    float vfa = w * v1 + (1 - w) * v2, vfb = w * v2 + (1 - w) * v1;
    float ra = rsqrtf(vfa + EPSV), rb = rsqrtf(vfb + EPSV);
    float y  = g1[e] * ((Y[row * EE + e]  - mfa) * ra) + b1[e];
    float y2 = g2[e] * ((Y2[row * EE + e] - mfb) * rb) + b2[e];
    Y[row * EE + e]  = y;
    Y2[row * EE + e] = y2;
    const int b = (int)(row / SSQ);
    atomicAdd(&rmY[b * EE + e],  y  * (1.0f / SSQ));
    atomicAdd(&rmY2[b * EE + e], y2 * (1.0f / SSQ));
}

__global__ __launch_bounds__(256)
void mmd_kernel(const float* __restrict__ rmY, const float* __restrict__ rmY2,
                float* __restrict__ loss)
{
    __shared__ float red[256];
    const int t = threadIdx.x;
    const int i = t >> 4, j = t & 15;   // 16x16 pairwise matrix, one pair per thread
    const float* ri = (i < 8) ? (rmY + i * EE) : (rmY2 + (i - 8) * EE);
    const float* rj = (j < 8) ? (rmY + j * EE) : (rmY2 + (j - 8) * EE);
    float d = 0.0f;
    for (int e = 0; e < EE; ++e) { float df = ri[e] - rj[e]; d += df * df; }
    red[t] = d;
    __syncthreads();
    for (int st = 128; st > 0; st >>= 1) { if (t < st) red[t] += red[t + st]; __syncthreads(); }
    float total = red[0];
    __syncthreads();
    float bw = total / (float)(16 * 16 - 16) / 4.0f;   // / kernel_mul^(5//2)
    float kern = 0.0f, mul = 1.0f;
    for (int q = 0; q < 5; ++q) { kern += __expf(-d / (bw * mul)); mul *= 2.0f; }
    float sign = ((i < 8) == (j < 8)) ? 1.0f : -1.0f;  // XX+YY vs XY+YX
    red[t] = sign * kern;
    __syncthreads();
    for (int st = 128; st > 0; st >>= 1) { if (t < st) red[t] += red[t + st]; __syncthreads(); }
    if (t == 0) loss[0] = red[0] / 64.0f;
}

// ---------------------------------------------------------------- launch
extern "C" void kernel_launch(void* const* d_in, const int* in_sizes, int n_in,
                              void* d_out, int out_size, void* d_ws, size_t ws_size,
                              hipStream_t stream)
{
    const float* x    = (const float*)d_in[0];
    const float* x2   = (const float*)d_in[1];
    const float* pde  = (const float*)d_in[2];
    const float* wi   = (const float*)d_in[3];
    const float* bi   = (const float*)d_in[4];
    const float* wo   = (const float*)d_in[5];
    const float* bo   = (const float*)d_in[6];
    const float* g1   = (const float*)d_in[7];
    const float* b1   = (const float*)d_in[8];
    const float* g2   = (const float*)d_in[9];
    const float* b2   = (const float*)d_in[10];
    const float* bnw  = (const float*)d_in[11];

    float* Y    = (float*)d_out;
    float* Y2   = Y + (size_t)MT * EE;
    float* loss = Y + 2 * (size_t)MT * EE;

    _Float16* w16 = (_Float16*)d_ws;
    size_t o = 0;
    _Float16* x16a = w16 + o; o += (size_t)MT * EE;
    _Float16* x16b = w16 + o; o += (size_t)MT * EE;
    _Float16* wi16 = w16 + o; o += (size_t)QN * EE;
    _Float16* wo16 = w16 + o; o += (size_t)EE * EE;
    _Float16* qkva = w16 + o; o += (size_t)MT * QN;
    _Float16* qkvb = w16 + o; o += (size_t)MT * QN;
    _Float16* atta = w16 + o; o += (size_t)MT * EE;
    _Float16* attb = w16 + o; o += (size_t)MT * EE;
    float* stats = (float*)(w16 + o);        // 1024 floats
    float* rmY   = stats + 1024;             // 8*256
    float* rmY2  = rmY + BB * EE;            // 8*256

    // f32 -> f16
    f32_to_f16_kernel<<<(MT * EE + 255) / 256, 256, 0, stream>>>(x,  x16a, MT * EE);
    f32_to_f16_kernel<<<(MT * EE + 255) / 256, 256, 0, stream>>>(x2, x16b, MT * EE);
    f32_to_f16_kernel<<<(QN * EE + 255) / 256, 256, 0, stream>>>(wi, wi16, QN * EE);
    f32_to_f16_kernel<<<(EE * EE + 255) / 256, 256, 0, stream>>>(wo, wo16, EE * EE);

    // QKV projection (fused bias + q-scale), f16 out
    dim3 gq(MT / 32, QN / 64);
    wmma_gemm_kernel<true><<<gq, 128, 0, stream>>>(x16a, wi16, bi, qkva, MT, QN, EE, QSCALE, EE);
    wmma_gemm_kernel<true><<<gq, 128, 0, stream>>>(x16b, wi16, bi, qkvb, MT, QN, EE, QSCALE, EE);

    // attention (flash-style, TDM double-buffered K/V staging)
    attn_kernel<<<BB * HH * (SSQ / 128), 256, 0, stream>>>(qkva, pde, atta);
    attn_kernel<<<BB * HH * (SSQ / 128), 256, 0, stream>>>(qkvb, pde, attb);

    // output projection, f32 out straight into d_out
    dim3 go(MT / 32, EE / 64);
    wmma_gemm_kernel<false><<<go, 128, 0, stream>>>(atta, wo16, bo, Y,  MT, EE, EE, 1.0f, 0);
    wmma_gemm_kernel<false><<<go, 128, 0, stream>>>(attb, wo16, bo, Y2, MT, EE, EE, 1.0f, 0);

    // dual BN + row means + MMD
    zero_kernel<<<(1024 + 2 * BB * EE + 255) / 256, 256, 0, stream>>>(stats, 1024 + 2 * BB * EE);
    bn_stats_kernel<<<MT / 32, 256, 0, stream>>>(Y, Y2, stats);
    bn_apply_kernel<<<MT, 256, 0, stream>>>(Y, Y2, stats, g1, b1, g2, b2, bnw, rmY, rmY2);
    mmd_kernel<<<1, 256, 0, stream>>>(rmY, rmY2, loss);
}
